// RelativeMHSA_46583215292591
// MI455X (gfx1250) — matrix operand order repaired and checked by reading, not compiled, not run on hardware
//
#include <hip/hip_runtime.h>

typedef __bf16 bf16_t;
typedef __attribute__((ext_vector_type(16))) __bf16 v16bf;
typedef __attribute__((ext_vector_type(8)))  __bf16 v8bf;
typedef __attribute__((ext_vector_type(8)))  float  v8f;

#define B_  16
#define T_  1024
#define C_  256
#define H_  8
#define HD_ 32
#define SCALE_ 0.17677669529663687f   // 1/sqrt(32)

static __device__ __forceinline__ bf16_t to_bf16(float f) { return (bf16_t)f; }

// Per-wave-private LDS round trips only need a compiler scheduling fence:
// DS ops from one wave are executed in-order by the LDS (DScnt, in-order),
// so RAW store->load within a wave is safe without s_barrier.
static __device__ __forceinline__ void wave_lds_fence() {
  __builtin_amdgcn_wave_barrier();
}

// ---------------------------------------------------------------------------
// Kernel 0: bulk f32 -> bf16 conversion (x, w_qkv, w_proj pre-pass).
// 8 elements / thread, b128 in / b128 out.
// ---------------------------------------------------------------------------
__global__ __launch_bounds__(256) void cvt_bf16_kernel(
    const float* __restrict__ in, bf16_t* __restrict__ out, int n)
{
  const int i = (blockIdx.x * 256 + threadIdx.x) * 8;
  if (i + 8 <= n) {
    const float4* p = (const float4*)(in + i);
    float4 a = p[0], b = p[1];
    v8bf o;
    o[0] = to_bf16(a.x); o[1] = to_bf16(a.y); o[2] = to_bf16(a.z); o[3] = to_bf16(a.w);
    o[4] = to_bf16(b.x); o[5] = to_bf16(b.y); o[6] = to_bf16(b.z); o[7] = to_bf16(b.w);
    *(v8bf*)(out + i) = o;
  }
}

// ---------------------------------------------------------------------------
// Kernel 1: QKV GEMM (bf16 in, f32 acc).  qkv[m,n] = x[m,:].w_qkv[n,:] + b[n]
// M = 16384, K = 256, N = 768.  Each wave: 16x64 tile (4 accumulators reusing
// one A fragment -> 4 WMMAs per A load).  Q pre-scaled; V stored transposed.
// ---------------------------------------------------------------------------
__global__ __launch_bounds__(256) void qkv_kernel(
    const bf16_t* __restrict__ xb, const bf16_t* __restrict__ wq,
    const float* __restrict__ bqkv,
    bf16_t* __restrict__ qB, bf16_t* __restrict__ kB, bf16_t* __restrict__ vT)
{
  const int lane  = threadIdx.x & 31;
  const int laneM = lane & 15;
  const int laneH = lane >> 4;
  const int wid   = blockIdx.x * 8 + (threadIdx.x >> 5);
  const int mtile = wid & 1023;     // 1024 tiles over M
  const int ngrp  = wid >> 10;      // 12 groups of 64 columns
  const int m     = mtile * 16 + laneM;

  const bf16_t* arow = xb + (size_t)m * C_;

  v8f acc[4];
  #pragma unroll
  for (int j = 0; j < 4; ++j) acc[j] = (v8f){0.f,0.f,0.f,0.f,0.f,0.f,0.f,0.f};

  #pragma unroll
  for (int k0 = 0; k0 < C_; k0 += 32) {
    // A fragment: per-lane K pattern {8*laneH..+7, 16+8*laneH..+7}
    const v8bf* pa = (const v8bf*)(arow + k0 + 8 * laneH);
    v8bf a0 = pa[0], a1 = pa[2];
    v16bf a;
    #pragma unroll
    for (int i = 0; i < 8; ++i) { a[i] = a0[i]; a[i + 8] = a1[i]; }
    #pragma unroll
    for (int j = 0; j < 4; ++j) {
      // B fragment: per-lane K pattern {8*laneH..+7, 16+8*laneH..+7}
      const int n = ngrp * 64 + j * 16 + laneM;
      const v8bf* pb = (const v8bf*)(wq + (size_t)n * C_ + k0 + 8 * laneH);
      v8bf b0 = pb[0], b1 = pb[2];
      v16bf bm;
      #pragma unroll
      for (int i = 0; i < 8; ++i) { bm[i] = b0[i]; bm[i + 8] = b1[i]; }
      acc[j] = __builtin_amdgcn_wmma_f32_16x16x32_bf16(false, a, false, bm,
                                                       (short)0, acc[j], false, false);
    }
  }

  const int which = (ngrp * 64) >> 8;          // 0=q, 1=k, 2=v (uniform)
  const int t0    = mtile * 16 + 8 * laneH;    // this lane's 8 rows
  const int bidx  = t0 >> 10;
  const int trow  = t0 & 1023;

  #pragma unroll
  for (int j = 0; j < 4; ++j) {
    const int n = ngrp * 64 + j * 16 + laneM;  // D column for this lane
    const int c = n & 255;
    const int h = c >> 5;
    const int d = c & 31;
    const float bias = bqkv[n];
    if (which == 2) {
      v8bf pk;
      #pragma unroll
      for (int r = 0; r < 8; ++r) pk[r] = to_bf16(acc[j][r] + bias);
      *(v8bf*)(vT + ((size_t)((bidx * H_ + h) * HD_ + d)) * T_ + trow) = pk;
    } else {
      const float sc = (which == 0) ? SCALE_ : 1.0f;
      bf16_t* dst = (which == 0 ? qB : kB)
                    + ((size_t)(bidx * H_ + h) * T_ + trow) * HD_ + d;
      #pragma unroll
      for (int r = 0; r < 8; ++r) dst[(size_t)r * HD_] = to_bf16((acc[j][r] + bias) * sc);
    }
  }
}

// ---------------------------------------------------------------------------
// Kernel 2: fused flash-style attention.  One wave owns a 16-row Q tile of one
// (b,h).  Per 32-key block: two QK^T WMMAs -> +bias -> online softmax
// (16-lane shfl_xor reductions, v_exp_f32) -> P through per-wave LDS slab
// (C->A relayout, guarded by wave-level fences only) -> two PV WMMAs.
// ---------------------------------------------------------------------------
__global__ __launch_bounds__(256) void attn_kernel(
    const bf16_t* __restrict__ qB, const bf16_t* __restrict__ kB,
    const bf16_t* __restrict__ vT, const float* __restrict__ bias_table,
    const int* __restrict__ rel_index, bf16_t* __restrict__ aoB)
{
  __shared__ __align__(16) bf16_t psh[8 * 16 * 32];   // 1KB per wave
  const int lane  = threadIdx.x & 31;
  const int laneM = lane & 15;
  const int laneH = lane >> 4;
  const int wv    = threadIdx.x >> 5;
  bf16_t* pbase   = psh + wv * (16 * 32);

  const int wid = blockIdx.x * 8 + wv;
  const int qt  = wid & 63;           // 64 q-tiles per head
  const int bh  = wid >> 6;           // 0..127
  const int b   = bh >> 3;
  const int h   = bh & 7;

  const size_t headbase = (size_t)(b * H_ + h) * T_ * HD_;

  // Q A-tile: K-dim = hd = 32 exactly -> one v16bf, loaded once.
  const int tq_lane = qt * 16 + laneM;
  const v8bf* qp = (const v8bf*)(qB + headbase + (size_t)tq_lane * HD_ + 8 * laneH);
  v8bf q0 = qp[0], q1 = qp[2];
  v16bf qa;
  #pragma unroll
  for (int i = 0; i < 8; ++i) { qa[i] = q0[i]; qa[i + 8] = q1[i]; }

  v8f o0 = {0.f,0.f,0.f,0.f,0.f,0.f,0.f,0.f};
  v8f o1 = {0.f,0.f,0.f,0.f,0.f,0.f,0.f,0.f};
  float mrow[8], lrow[8];
  #pragma unroll
  for (int r = 0; r < 8; ++r) { mrow[r] = -1e30f; lrow[r] = 0.f; }

  const int tq0 = qt * 16 + 8 * laneH;   // stats rows for this lane
  const bf16_t* vbase = vT + (size_t)(b * H_ + h) * HD_ * T_;

  for (int kt = 0; kt < 32; ++kt) {      // 32 blocks of 32 keys
    v8f S[2];
    #pragma unroll
    for (int sub = 0; sub < 2; ++sub) {
      const int kb  = kt * 32 + sub * 16;
      const int key = kb + laneM;        // B column = key index
      const v8bf* kp = (const v8bf*)(kB + headbase + (size_t)key * HD_ + 8 * laneH);
      v8bf k0v = kp[0], k1v = kp[2];
      v16bf kbm;
      #pragma unroll
      for (int i = 0; i < 8; ++i) { kbm[i] = k0v[i]; kbm[i + 8] = k1v[i]; }
      v8f z = {0.f,0.f,0.f,0.f,0.f,0.f,0.f,0.f};
      v8f s = __builtin_amdgcn_wmma_f32_16x16x32_bf16(false, qa, false, kbm,
                                                      (short)0, z, false, false);
      // relative-position bias gather (rel_index + bias_table are L2-resident)
      #pragma unroll
      for (int r = 0; r < 8; ++r) {
        const int tq  = tq0 + r;
        const int idx = rel_index[(size_t)tq * T_ + kb + laneM];
        s[r] += bias_table[idx * H_ + h];
      }
      S[sub] = s;
    }

    // online softmax over the 32-key block (rows live across 16-lane halves)
    #pragma unroll
    for (int r = 0; r < 8; ++r) {
      float v = fmaxf(S[0][r], S[1][r]);
      v = fmaxf(v, __shfl_xor(v, 1, 32));
      v = fmaxf(v, __shfl_xor(v, 2, 32));
      v = fmaxf(v, __shfl_xor(v, 4, 32));
      v = fmaxf(v, __shfl_xor(v, 8, 32));
      const float mn    = fmaxf(mrow[r], v);
      const float alpha = __expf(mrow[r] - mn);
      const float p0 = __expf(S[0][r] - mn);
      const float p1 = __expf(S[1][r] - mn);
      S[0][r] = p0; S[1][r] = p1;
      float ps = p0 + p1;
      ps += __shfl_xor(ps, 1, 32);
      ps += __shfl_xor(ps, 2, 32);
      ps += __shfl_xor(ps, 4, 32);
      ps += __shfl_xor(ps, 8, 32);
      lrow[r] = lrow[r] * alpha + ps;
      mrow[r] = mn;
      o0[r] *= alpha;
      o1[r] *= alpha;
    }

    // P: C-layout (lane=key col, vgpr=row) -> LDS [16 q][32 k] row-major
    #pragma unroll
    for (int sub = 0; sub < 2; ++sub)
      #pragma unroll
      for (int r = 0; r < 8; ++r)
        pbase[(8 * laneH + r) * 32 + sub * 16 + laneM] = to_bf16(S[sub][r]);
    wave_lds_fence();   // keep DS loads after DS stores (HW is in-order per wave)

    // reload P in A-layout: lane row = laneM, K chunks at 8*laneH and +16
    const v8bf* pp = (const v8bf*)(pbase + laneM * 32 + 8 * laneH);
    v8bf p0 = pp[0], p1 = pp[2];
    v16bf pa;
    #pragma unroll
    for (int i = 0; i < 8; ++i) { pa[i] = p0[i]; pa[i + 8] = p1[i]; }
    wave_lds_fence();   // keep next iteration's stores after these loads

    // V B-tiles: K-dim = key (contiguous in vT), N = d (two halves of hd=32)
    #pragma unroll
    for (int tile = 0; tile < 2; ++tile) {
      const int d = tile * 16 + laneM;
      const v8bf* vp = (const v8bf*)(vbase + (size_t)d * T_ + kt * 32 + 8 * laneH);
      v8bf vv0 = vp[0], vv1 = vp[2];
      v16bf vb;
      #pragma unroll
      for (int i = 0; i < 8; ++i) { vb[i] = vv0[i]; vb[i + 8] = vv1[i]; }
      if (tile == 0)
        o0 = __builtin_amdgcn_wmma_f32_16x16x32_bf16(false, pa, false, vb,
                                                     (short)0, o0, false, false);
      else
        o1 = __builtin_amdgcn_wmma_f32_16x16x32_bf16(false, pa, false, vb,
                                                     (short)0, o1, false, false);
    }
  }

  // epilogue: normalize and store attn output as bf16 [B, T, C]
  #pragma unroll
  for (int r = 0; r < 8; ++r) {
    const float inv = 1.0f / lrow[r];
    const int t = tq0 + r;
    bf16_t* dst = aoB + ((size_t)(b * T_ + t)) * C_ + h * HD_;
    dst[laneM]      = to_bf16(o0[r] * inv);
    dst[16 + laneM] = to_bf16(o1[r] * inv);
  }
}

// ---------------------------------------------------------------------------
// Kernel 3: output projection.  out[m,n] = ao[m,:].w_proj[n,:] + b_proj[n]
// M = 16384, K = 256, N = 256.  16x64 tile per wave, all-bf16 operands.
// ---------------------------------------------------------------------------
__global__ __launch_bounds__(256) void proj_kernel(
    const bf16_t* __restrict__ aoB, const bf16_t* __restrict__ wp,
    const float* __restrict__ bproj, float* __restrict__ out)
{
  const int lane  = threadIdx.x & 31;
  const int laneM = lane & 15;
  const int laneH = lane >> 4;
  const int wid   = blockIdx.x * 8 + (threadIdx.x >> 5);
  const int mtile = wid & 1023;
  const int ngrp  = wid >> 10;          // 4 groups of 64 columns
  const int m     = mtile * 16 + laneM;

  const bf16_t* arow = aoB + (size_t)m * C_;

  v8f acc[4];
  #pragma unroll
  for (int j = 0; j < 4; ++j) acc[j] = (v8f){0.f,0.f,0.f,0.f,0.f,0.f,0.f,0.f};

  #pragma unroll
  for (int k0 = 0; k0 < C_; k0 += 32) {
    const v8bf* pa = (const v8bf*)(arow + k0 + 8 * laneH);
    v8bf a0 = pa[0], a1 = pa[2];
    v16bf a;
    #pragma unroll
    for (int i = 0; i < 8; ++i) { a[i] = a0[i]; a[i + 8] = a1[i]; }
    #pragma unroll
    for (int j = 0; j < 4; ++j) {
      const int n = ngrp * 64 + j * 16 + laneM;
      const v8bf* pb = (const v8bf*)(wp + (size_t)n * C_ + k0 + 8 * laneH);
      v8bf b0 = pb[0], b1 = pb[2];
      v16bf bm;
      #pragma unroll
      for (int i = 0; i < 8; ++i) { bm[i] = b0[i]; bm[i + 8] = b1[i]; }
      acc[j] = __builtin_amdgcn_wmma_f32_16x16x32_bf16(false, a, false, bm,
                                                       (short)0, acc[j], false, false);
    }
  }

  const int t0 = mtile * 16 + 8 * laneH;
  #pragma unroll
  for (int j = 0; j < 4; ++j) {
    const int n = ngrp * 64 + j * 16 + laneM;
    const float bias = bproj[n];
    #pragma unroll
    for (int r = 0; r < 8; ++r)
      out[(size_t)(t0 + r) * C_ + n] = acc[j][r] + bias;
  }
}

// ---------------------------------------------------------------------------
extern "C" void kernel_launch(void* const* d_in, const int* in_sizes, int n_in,
                              void* d_out, int out_size, void* d_ws, size_t ws_size,
                              hipStream_t stream) {
  const float* x     = (const float*)d_in[0];
  const float* wqkv  = (const float*)d_in[1];
  const float* bqkv  = (const float*)d_in[2];
  const float* wproj = (const float*)d_in[3];
  const float* bproj = (const float*)d_in[4];
  const float* btab  = (const float*)d_in[5];
  const int*   ridx  = (const int*)d_in[6];

  char* ws = (char*)d_ws;
  bf16_t* qB   = (bf16_t*)(ws);                            // 8 MB [B,H,T,hd]
  bf16_t* kB   = (bf16_t*)(ws + (size_t)( 8u << 20));      // 8 MB [B,H,T,hd]
  bf16_t* vT   = (bf16_t*)(ws + (size_t)(16u << 20));      // 8 MB [B,H,hd,T]
  bf16_t* aoB  = (bf16_t*)(ws + (size_t)(24u << 20));      // 8 MB [B,T,C]
  bf16_t* xb   = (bf16_t*)(ws + (size_t)(32u << 20));      // 8 MB [B*T,C]
  bf16_t* wqb  = (bf16_t*)(ws + (size_t)(40u << 20));      // 384 KB [3C,C]
  bf16_t* wpb  = (bf16_t*)(ws + (size_t)(40u << 20) + (512u << 10)); // 128 KB [C,C]

  const int NX  = B_ * T_ * C_;      // 4194304
  const int NWQ = 3 * C_ * C_;       // 196608
  const int NWP = C_ * C_;           // 65536
  cvt_bf16_kernel<<<NX  / 2048, 256, 0, stream>>>(x,     xb,  NX);
  cvt_bf16_kernel<<<NWQ / 2048, 256, 0, stream>>>(wqkv,  wqb, NWQ);
  cvt_bf16_kernel<<<NWP / 2048, 256, 0, stream>>>(wproj, wpb, NWP);

  qkv_kernel <<<1536, 256, 0, stream>>>(xb, wqb, bqkv, qB, kB, vT);
  attn_kernel<<<1024, 256, 0, stream>>>(qB, kB, vT, btab, ridx, aoB);
  proj_kernel<<< 512, 256, 0, stream>>>(aoB, wpb, bproj, (float*)d_out);
}
